// SVDBlock_6786048327892
// MI455X (gfx1250) — matrix operand order repaired and checked by reading, not compiled, not run on hardware
//
#include <hip/hip_runtime.h>

typedef _Float16 f16;
typedef __attribute__((ext_vector_type(16))) _Float16 v16h;
typedef __attribute__((ext_vector_type(8)))  float    v8f;
typedef __attribute__((ext_vector_type(2)))  _Float16 h2;
typedef __attribute__((ext_vector_type(8)))  _Float16 h8;
typedef __attribute__((ext_vector_type(4)))  int      v4i;

#define DEV static __device__ __forceinline__

constexpr int Bb = 16, Mm = 1024, DM = 768, H = 12, R = 32;
constexpr int DFF = 3072, RFF = 384, RWO = 768, DH = 64;
constexpr int NTOK = Bb * Mm;  // 16384 tokens
constexpr int C3 = 3 * DM;     // 2304 fused QKV columns

// ---- gfx1250 async global->LDS path (guarded; falls back to sync copies) ---
#if defined(__has_builtin)
#if __has_builtin(__builtin_amdgcn_global_load_async_to_lds_b128)
#define HAVE_ASYNC 1
#else
#define HAVE_ASYNC 0
#endif
#if __has_builtin(__builtin_amdgcn_s_wait_asynccnt)
#define HAVE_WAITA 1
#else
#define HAVE_WAITA 0
#endif
#else
#define HAVE_ASYNC 0
#define HAVE_WAITA 0
#endif

typedef __attribute__((address_space(1))) v4i* gas_p;  // global (AS1) 16B vec
typedef __attribute__((address_space(3))) v4i* las_p;  // LDS (AS3) 16B vec

DEV void cp_lds16(f16* lds_dst, const f16* gsrc) {
#if HAVE_ASYNC
  __builtin_amdgcn_global_load_async_to_lds_b128((gas_p)gsrc, (las_p)lds_dst, 0, 0);
#else
  *(h8*)lds_dst = *(const h8*)gsrc;
#endif
}

template <int N>
DEV void wait_async() {
#if HAVE_ASYNC
#if HAVE_WAITA
  __builtin_amdgcn_s_wait_asynccnt(N);
#else
  asm volatile("s_wait_asynccnt %0" ::"i"(N));
#endif
#endif
}

// ---- CDNA5 wave32 WMMA fragment index maps (ISA 7.12.2, 16-bit A/B) -------
// A 16x32: lane m = lane%16, hs = lane/16; VGPR j holds k-pair below
DEV int kmapA(int j, int hs) { return (j < 4) ? (8 * hs + 2 * j) : (16 + 8 * hs + 2 * (j - 4)); }
// B 32x16: lane n = lane%16; VGPR j holds k-pair below (same k order as A)
DEV int kmapB(int j, int hs) { return (j < 4) ? (8 * hs + 2 * j) : (16 + 8 * hs + 2 * (j - 4)); }

DEV v8f wmma32(v16h a, v16h b, v8f c) {
  return __builtin_amdgcn_wmma_f32_16x16x32_f16(false, a, false, b, (short)0, c, false, false);
}

// ============================================================================
// f16 GEMM, f32 accumulate: C = A[M,K] * Bt[N,K]^T + bias (opt exact GELU).
// Block tile 128x128, 8 waves (wave tile 32x64), k-step 32, double-buffered
// LDS staging via async global->LDS copies (ASYNCcnt).
// ============================================================================
template <int ACT>
__global__ __launch_bounds__(256) void gemm_f16(
    const f16* __restrict__ A, int lda, const f16* __restrict__ Bt, int ldbt,
    const float* __restrict__ bias, f16* __restrict__ outH,
    float* __restrict__ outF, int Mrows, int Ncols, int K) {
  __shared__ __align__(16) f16 As[2][128][40];  // [buf][m][k] (+8 pad)
  __shared__ __align__(16) f16 Bs[2][128][40];  // [buf][n][k] (+8 pad)
  const int t = threadIdx.x;
  const int lane = t & 31, wave = t >> 5;
  const int ln16 = lane & 15, hs = lane >> 4;
  const int wm = wave & 3, wn = wave >> 2;  // 4 waves over M, 2 over N
  const int bM = blockIdx.y * 128, bN = blockIdx.x * 128;
  const int nk = K >> 5;
  (void)Mrows;

  v8f acc[2][4];
#pragma unroll
  for (int i = 0; i < 2; i++)
#pragma unroll
    for (int j = 0; j < 4; j++)
#pragma unroll
      for (int e = 0; e < 8; e++) acc[i][j][e] = 0.0f;

  // stage tile s into LDS buffer `buf` (4 async ops per wave: 2 A + 2 B)
  auto stage = [&](int s, int buf) {
    const int k0 = s * 32;
#pragma unroll
    for (int i = 0; i < 2; i++) {
      int f = t + 256 * i;  // 0..511
      int r = f >> 2, c4 = f & 3;
      cp_lds16(&As[buf][r][c4 * 8], &A[(size_t)(bM + r) * lda + k0 + c4 * 8]);
    }
#pragma unroll
    for (int i = 0; i < 2; i++) {
      int f = t + 256 * i;
      int r = f >> 2, c4 = f & 3;
      cp_lds16(&Bs[buf][r][c4 * 8], &Bt[(size_t)(bN + r) * ldbt + k0 + c4 * 8]);
    }
  };

  stage(0, 0);
  for (int s = 0; s < nk; s++) {
    const int buf = s & 1;
    __syncthreads();  // all waves done reading buf^1 -> safe to refill it
    if (s + 1 < nk) {
      stage(s + 1, buf ^ 1);
      wait_async<4>();  // stage s retired (async loads complete in order)
    } else {
      wait_async<0>();
    }
    __syncthreads();  // stage-s data from every wave visible in LDS

    v16h af[2], bf[4];
#pragma unroll
    for (int mt = 0; mt < 2; mt++) {
      int row = wm * 32 + mt * 16 + ln16;
#pragma unroll
      for (int j = 0; j < 8; j++) {
        h2 pr = *(const h2*)&As[buf][row][kmapA(j, hs)];
        af[mt][2 * j] = pr.x;
        af[mt][2 * j + 1] = pr.y;
      }
    }
#pragma unroll
    for (int nb = 0; nb < 4; nb++) {
      int col = wn * 64 + nb * 16 + ln16;
#pragma unroll
      for (int j = 0; j < 8; j++) {
        h2 pr = *(const h2*)&Bs[buf][col][kmapB(j, hs)];
        bf[nb][2 * j] = pr.x;
        bf[nb][2 * j + 1] = pr.y;
      }
    }
#pragma unroll
    for (int mt = 0; mt < 2; mt++)
#pragma unroll
      for (int nb = 0; nb < 4; nb++) acc[mt][nb] = wmma32(af[mt], bf[nb], acc[mt][nb]);
  }

// epilogue: C layout VGPR r -> rows r (lanes 0-15) / 8+r (lanes 16-31)
#pragma unroll
  for (int nb = 0; nb < 4; nb++) {
    int col = bN + wn * 64 + nb * 16 + ln16;
    float bvl = bias ? bias[col] : 0.0f;
#pragma unroll
    for (int mt = 0; mt < 2; mt++) {
#pragma unroll
      for (int r = 0; r < 8; r++) {
        int row = bM + wm * 32 + mt * 16 + hs * 8 + r;
        float v = acc[mt][nb][r] + bvl;
        if (ACT == 1) v = 0.5f * v * (1.0f + erff(v * 0.70710678118654752f));  // exact GELU
        if (outF) outF[(size_t)row * Ncols + col] = v;
        if (outH) outH[(size_t)row * Ncols + col] = (f16)v;
      }
    }
  }
}

// ============================================================================
// Flash attention: one WG per (b,h,64-query tile); 4 waves, 16 q-rows/wave.
// qkv: [NTOK, 2304] f16 (Q|K|V, head-major cols). Online softmax in f32.
// V staged transposed so P*V B-fragments read contiguous k-pairs.
// ============================================================================
__global__ __launch_bounds__(128) void attn_kernel(const f16* __restrict__ qkv,
                                                   const int* __restrict__ mask,
                                                   f16* __restrict__ attn) {
  __shared__ __align__(16) f16 Ks[64][72];  // [key][d]
  __shared__ __align__(16) f16 Vt[64][72];  // [d][key] (transposed)
  __shared__ __align__(16) f16 Ps[4][16][72];
  const int t = threadIdx.x, lane = t & 31, wave = t >> 5;
  const int ln16 = lane & 15, hs = lane >> 4;
  const int qt = blockIdx.x, h = blockIdx.y, b = blockIdx.z;
  const size_t rs = (size_t)C3;
  const int qrow0 = b * Mm + qt * 64;

  // Q fragments (A layout), loaded once
  v16h qA[2];
  {
    const f16* qp = qkv + (size_t)(qrow0 + wave * 16 + ln16) * rs + h * DH;
#pragma unroll
    for (int ks = 0; ks < 2; ks++)
#pragma unroll
      for (int j = 0; j < 8; j++) {
        h2 pr = *(const h2*)&qp[32 * ks + kmapA(j, hs)];
        qA[ks][2 * j] = pr.x;
        qA[ks][2 * j + 1] = pr.y;
      }
  }

  float mrow[8], lrow[8];
  v8f o[4];
#pragma unroll
  for (int r = 0; r < 8; r++) { mrow[r] = -3.0e38f; lrow[r] = 0.0f; }
#pragma unroll
  for (int nb = 0; nb < 4; nb++)
#pragma unroll
    for (int e = 0; e < 8; e++) o[nb][e] = 0.0f;

  for (int kt = 0; kt < 16; kt++) {
    const int key0 = kt * 64;
// stage K (direct) and V (transposed) tiles
#pragma unroll
    for (int i = 0; i < 4; i++) {
      int f = t + 128 * i;  // 0..511
      int r = f >> 3, c8 = f & 7;
      const f16* src = qkv + (size_t)(b * Mm + key0 + r) * rs + DM + h * DH + c8 * 8;
      *(h8*)&Ks[r][c8 * 8] = *(const h8*)src;
      h8 vv = *(const h8*)(src + DM);
#pragma unroll
      for (int e = 0; e < 8; e++) Vt[c8 * 8 + e][r] = vv[e];
    }
    __syncthreads();

    // S = Q * K^T  (B fragment: lane n = key, k = d contiguous in Ks rows)
    v8f s[4];
#pragma unroll
    for (int nb = 0; nb < 4; nb++)
#pragma unroll
      for (int e = 0; e < 8; e++) s[nb][e] = 0.0f;
#pragma unroll
    for (int ks = 0; ks < 2; ks++) {
#pragma unroll
      for (int nb = 0; nb < 4; nb++) {
        v16h bf;
#pragma unroll
        for (int j = 0; j < 8; j++) {
          h2 pr = *(const h2*)&Ks[nb * 16 + ln16][32 * ks + kmapB(j, hs)];
          bf[2 * j] = pr.x;
          bf[2 * j + 1] = pr.y;
        }
        s[nb] = wmma32(qA[ks], bf, s[nb]);
      }
    }
    // scale + key padding mask
    bool mk[4];
#pragma unroll
    for (int nb = 0; nb < 4; nb++) mk[nb] = mask[b * Mm + key0 + nb * 16 + ln16] != 0;
#pragma unroll
    for (int nb = 0; nb < 4; nb++)
#pragma unroll
      for (int r = 0; r < 8; r++) s[nb][r] = mk[nb] ? s[nb][r] * 0.125f : -1.0e9f;

// online softmax: lane owns rows hs*8+r over its 16-lane half (C layout)
#pragma unroll
    for (int r = 0; r < 8; r++) {
      float tm = fmaxf(fmaxf(s[0][r], s[1][r]), fmaxf(s[2][r], s[3][r]));
#pragma unroll
      for (int off = 1; off < 16; off <<= 1) tm = fmaxf(tm, __shfl_xor(tm, off, 32));
      float mnew = fmaxf(mrow[r], tm);
      float alpha = __expf(mrow[r] - mnew);
      float psum = 0.0f;
#pragma unroll
      for (int nb = 0; nb < 4; nb++) {
        float p = __expf(s[nb][r] - mnew);
        psum += p;
        Ps[wave][hs * 8 + r][nb * 16 + ln16] = (f16)p;
      }
#pragma unroll
      for (int off = 1; off < 16; off <<= 1) psum += __shfl_xor(psum, off, 32);
      lrow[r] = lrow[r] * alpha + psum;
      mrow[r] = mnew;
#pragma unroll
      for (int nb = 0; nb < 4; nb++) o[nb][r] *= alpha;
    }

// O += P * V  (A from per-wave Ps; B from Vt: lane n = d row, k = key contig)
#pragma unroll
    for (int ks = 0; ks < 2; ks++) {
      v16h pa;
#pragma unroll
      for (int j = 0; j < 8; j++) {
        h2 pr = *(const h2*)&Ps[wave][ln16][32 * ks + kmapA(j, hs)];
        pa[2 * j] = pr.x;
        pa[2 * j + 1] = pr.y;
      }
#pragma unroll
      for (int nbd = 0; nbd < 4; nbd++) {
        v16h vb;
#pragma unroll
        for (int j = 0; j < 8; j++) {
          h2 pr = *(const h2*)&Vt[nbd * 16 + ln16][32 * ks + kmapB(j, hs)];
          vb[2 * j] = pr.x;
          vb[2 * j + 1] = pr.y;
        }
        o[nbd] = wmma32(pa, vb, o[nbd]);
      }
    }
    __syncthreads();
  }

// write [B,M,H,DH] == [NTOK, DM] head-major, f16 for next GEMM
#pragma unroll
  for (int nb = 0; nb < 4; nb++)
#pragma unroll
    for (int r = 0; r < 8; r++) {
      int grow = qrow0 + wave * 16 + hs * 8 + r;
      int col = h * DH + nb * 16 + ln16;
      float denom = fmaxf(lrow[r], 1e-30f);
      attn[(size_t)grow * DM + col] = (f16)(o[nb][r] / denom);
    }
}

// ============================================================================
// Residual + LayerNorm over DM=768 (one row per WG); opt f16 copy for chaining
// ============================================================================
template <bool WH>
__global__ __launch_bounds__(256) void ln_res(const float* __restrict__ xa,
                                              const float* __restrict__ xb,
                                              const float* __restrict__ g,
                                              const float* __restrict__ be,
                                              float* __restrict__ outF,
                                              f16* __restrict__ outH) {
  __shared__ float s1[256], s2[256];
  const int row = blockIdx.x, t = threadIdx.x;
  float v[3], su = 0.0f, sq = 0.0f;
#pragma unroll
  for (int i = 0; i < 3; i++) {
    int c = t + 256 * i;
    float x = xa[(size_t)row * DM + c] + xb[(size_t)row * DM + c];
    v[i] = x; su += x; sq += x * x;
  }
  s1[t] = su; s2[t] = sq;
  __syncthreads();
  for (int off = 128; off > 0; off >>= 1) {
    if (t < off) { s1[t] += s1[t + off]; s2[t] += s2[t + off]; }
    __syncthreads();
  }
  float mu = s1[0] * (1.0f / DM);
  float var = s2[0] * (1.0f / DM) - mu * mu;
  float rstd = rsqrtf(var + 1e-12f);
#pragma unroll
  for (int i = 0; i < 3; i++) {
    int c = t + 256 * i;
    float y = (v[i] - mu) * rstd * g[c] + be[c];
    outF[(size_t)row * DM + c] = y;
    if (WH) outH[(size_t)row * DM + c] = (f16)y;
  }
}

// ---- weight fusion (low-rank pairs -> dense f16, stored TRANSPOSED [N][K]) -
__global__ void fuse_qkv(const float* Pq, const float* Vq, const float* Pk,
                         const float* Vk, const float* Pv, const float* Vv,
                         f16* __restrict__ Wt) {
  int idx = blockIdx.x * blockDim.x + threadIdx.x;
  if (idx >= DM * C3) return;
  int d = idx / C3, c = idx % C3;  // d = K index, c = N index
  int s = c / DM, rem = c % DM;
  int h = rem / DH, e = rem % DH;
  const float* P = (s == 0) ? Pq : (s == 1) ? Pk : Pv;
  const float* V = (s == 0) ? Vq : (s == 1) ? Vk : Vv;
  float acc = 0.0f;
  for (int r = 0; r < R; r++)
    acc += P[((size_t)h * DM + d) * R + r] * V[((size_t)h * R + r) * DH + e];
  Wt[(size_t)c * DM + d] = (f16)acc;  // [N=2304][K=768]
}

__global__ void fuse_bias(const float* bq, const float* bk, const float* bv,
                          float* bias) {
  int c = blockIdx.x * blockDim.x + threadIdx.x;
  if (c >= C3) return;
  int s = c / DM, rem = c % DM;
  bias[c] = ((s == 0) ? bq : (s == 1) ? bk : bv)[rem];
}

// Wt[n][k] = sum_r A[k][r] * Bm[r][n]
__global__ void fuse2t(const float* A, const float* Bm, f16* __restrict__ Wt,
                       int Kdim, int Nc, int Rk) {
  int idx = blockIdx.x * blockDim.x + threadIdx.x;
  if (idx >= Kdim * Nc) return;
  int k = idx / Nc, n = idx % Nc;
  float acc = 0.0f;
  for (int rr = 0; rr < Rk; rr++) acc += A[(size_t)k * Rk + rr] * Bm[(size_t)rr * Nc + n];
  Wt[(size_t)n * Kdim + k] = (f16)acc;
}

__global__ void cvt_f16(const float* __restrict__ x, f16* __restrict__ y, int n) {
  int i = blockIdx.x * blockDim.x + threadIdx.x;
  if (i < n) y[i] = (f16)x[i];
}

// ============================================================================
extern "C" void kernel_launch(void* const* d_in, const int* in_sizes, int n_in,
                              void* d_out, int out_size, void* d_ws,
                              size_t ws_size, hipStream_t stream) {
  (void)in_sizes; (void)n_in; (void)out_size; (void)ws_size;
  const float* x  = (const float*)d_in[0];
  const int* mask = (const int*)d_in[1];
  const float *Pq = (const float*)d_in[2],  *Vq = (const float*)d_in[3],  *bq = (const float*)d_in[4];
  const float *Pk = (const float*)d_in[5],  *Vk = (const float*)d_in[6],  *bk = (const float*)d_in[7];
  const float *Pv = (const float*)d_in[8],  *Vv = (const float*)d_in[9],  *bv = (const float*)d_in[10];
  const float *Uo = (const float*)d_in[11], *Vo = (const float*)d_in[12], *bo = (const float*)d_in[13];
  const float *U1 = (const float*)d_in[14], *V1 = (const float*)d_in[15], *b1 = (const float*)d_in[16];
  const float *U2 = (const float*)d_in[17], *V2 = (const float*)d_in[18], *b2 = (const float*)d_in[19];
  const float *g1 = (const float*)d_in[20], *be1 = (const float*)d_in[21];
  const float *g2 = (const float*)d_in[22], *be2 = (const float*)d_in[23];

  char* ws = (char*)d_ws;
  size_t off = 0;
  auto alloc = [&](size_t bytes) {
    void* p = ws + off;
    off = (off + bytes + 255) & ~(size_t)255;
    return p;
  };
  f16* WQKVt  = (f16*)alloc((size_t)DM * C3 * 2);    // [2304][768]
  float* bqkv = (float*)alloc((size_t)C3 * 4);
  f16* Wot    = (f16*)alloc((size_t)DM * DM * 2);    // [768][768]
  f16* W1t    = (f16*)alloc((size_t)DM * DFF * 2);   // [3072][768]
  f16* W2t    = (f16*)alloc((size_t)DFF * DM * 2);   // [768][3072]
  f16* Xh     = (f16*)alloc((size_t)NTOK * DM * 2);  // later reused as x1h
  f16* QKVh   = (f16*)alloc((size_t)NTOK * C3 * 2);
  f16* Attnh  = (f16*)alloc((size_t)NTOK * DM * 2);  // contiguous after QKVh
  f16* midh   = QKVh;  // aliases QKVh+Attnh region == NTOK*DFF halves exactly
  float* Yf   = (float*)alloc((size_t)NTOK * DM * 4);
  float* x1f  = (float*)alloc((size_t)NTOK * DM * 4);

  // 1) one-time conversions / fused-transposed weights
  cvt_f16<<<(NTOK * DM + 255) / 256, 256, 0, stream>>>(x, Xh, NTOK * DM);
  fuse_qkv<<<(DM * C3 + 255) / 256, 256, 0, stream>>>(Pq, Vq, Pk, Vk, Pv, Vv, WQKVt);
  fuse_bias<<<(C3 + 255) / 256, 256, 0, stream>>>(bq, bk, bv, bqkv);
  fuse2t<<<(DM * DM + 255) / 256, 256, 0, stream>>>(Uo, Vo, Wot, DM, DM, RWO);
  fuse2t<<<(DM * DFF + 255) / 256, 256, 0, stream>>>(U1, V1, W1t, DM, DFF, RFF);
  fuse2t<<<(DFF * DM + 255) / 256, 256, 0, stream>>>(U2, V2, W2t, DFF, DM, RFF);

  // 2) fused QKV projection: [16384,768] x [768,2304]
  gemm_f16<0><<<dim3(C3 / 128, NTOK / 128), 256, 0, stream>>>(
      Xh, DM, WQKVt, DM, bqkv, QKVh, nullptr, NTOK, C3, DM);

  // 3) flash attention per (b, h, 64-query tile)
  attn_kernel<<<dim3(Mm / 64, H, Bb), 128, 0, stream>>>(QKVh, mask, Attnh);

  // 4) attention output projection (fused Uo*Vo) -> f32
  gemm_f16<0><<<dim3(DM / 128, NTOK / 128), 256, 0, stream>>>(
      Attnh, DM, Wot, DM, bo, nullptr, Yf, NTOK, DM, DM);

  // 5) x1 = LN(x + attn_out); f32 for residual, f16 for FFN GEMM
  ln_res<true><<<NTOK, 256, 0, stream>>>(x, Yf, g1, be1, x1f, Xh);

  // 6) FFN up (fused U1*V1) with exact GELU epilogue
  gemm_f16<1><<<dim3(DFF / 128, NTOK / 128), 256, 0, stream>>>(
      Xh, DM, W1t, DM, b1, midh, nullptr, NTOK, DFF, DM);

  // 7) FFN down (fused U2*V2) -> f32
  gemm_f16<0><<<dim3(DM / 128, NTOK / 128), 256, 0, stream>>>(
      midh, DFF, W2t, DFF, b2, nullptr, Yf, NTOK, DM, DFF);

  // 8) out = LN(x1 + ffn)
  ln_res<false><<<NTOK, 256, 0, stream>>>(x1f, Yf, g2, be2, (float*)d_out, nullptr);
}
